// PFGAT_11562051961041
// MI455X (gfx1250) — compile-verified, matrix operand order checked by reading, not run on hardware
//
#include <hip/hip_runtime.h>
#include <hip/hip_bf16.h>
#include <math.h>
#include <stdint.h>

typedef __attribute__((ext_vector_type(16))) _Float16 v16h;
typedef __attribute__((ext_vector_type(8)))  float    v8f;

#define B_     4
#define T_     8
#define N_     2048
#define FIN_   16
#define HID_   128
#define ROWS_  (B_*T_*N_)   /* 65536 */
#define NODES_ (B_*N_)      /* 8192  */
#define E_     32768

// ---------------------------------------------------------------- WMMA helpers
__device__ __forceinline__ v8f wmma16(v16h a, v16h b, v8f c) {
  return __builtin_amdgcn_wmma_f32_16x16x32_f16(false, a, false, b, (short)0, c, false, false);
}
__device__ __forceinline__ v8f zero8() {
  v8f z;
#pragma unroll
  for (int i = 0; i < 8; ++i) z[i] = 0.f;
  return z;
}
// A fragment from row-major [16][ld] tile: element (m,k)
__device__ __forceinline__ v16h load_a(const _Float16* t, int ld) {
  int lane = threadIdx.x & 31;
  int m = lane & 15, khi = (lane >> 4) << 3;
  v16h a;
#pragma unroll
  for (int i = 0; i < 16; ++i) { int k = ((i >> 3) << 4) + khi + (i & 7); a[i] = t[m * ld + k]; }
  return a;
}
// B fragment from [k][n] row-major tile
__device__ __forceinline__ v16h load_b(const _Float16* t, int ld) {
  int lane = threadIdx.x & 31;
  int n = lane & 15, khi = (lane >> 4) << 3;
  v16h b;
#pragma unroll
  for (int i = 0; i < 16; ++i) { int k = ((i >> 3) << 4) + khi + (i & 7); b[i] = t[k * ld + n]; }
  return b;
}
// B fragment from [n][k] row-major tile (transposed access, for Q*K^T)
__device__ __forceinline__ v16h load_bt(const _Float16* t, int ld) {
  int lane = threadIdx.x & 31;
  int n = lane & 15, khi = (lane >> 4) << 3;
  v16h b;
#pragma unroll
  for (int i = 0; i < 16; ++i) { int k = ((i >> 3) << 4) + khi + (i & 7); b[i] = t[n * ld + k]; }
  return b;
}

// gfx1250 async direct global->LDS copy (16B per lane), tracked by ASYNCcnt
__device__ __forceinline__ void async_cp16(unsigned lds_addr, unsigned long long gaddr) {
  asm volatile("global_load_async_to_lds_b128 %0, %1, off"
               :: "v"(lds_addr), "v"(gaddr) : "memory");
}
#define WAIT_ASYNC(lit) asm volatile("s_wait_asynccnt " lit ::: "memory")

// sortable-uint mapping for float atomic max
__device__ __forceinline__ unsigned f2ord(float f) {
  unsigned u = __float_as_uint(f);
  return (u & 0x80000000u) ? ~u : (u | 0x80000000u);
}
__device__ __forceinline__ float ord2f(unsigned u) {
  return (u & 0x80000000u) ? __uint_as_float(u & 0x7FFFFFFFu) : __uint_as_float(~u);
}

// ---------------------------------------------------------------- small utils
__global__ void cvt_f32_f16_k(const float* __restrict__ s, _Float16* __restrict__ d, int n) {
  int t = blockIdx.x * blockDim.x + threadIdx.x;
  if (t < n) d[t] = (_Float16)s[t];
}
__global__ void fill_u32_k(unsigned* p, unsigned v, int n) {
  int t = blockIdx.x * blockDim.x + threadIdx.x;
  if (t < n) p[t] = v;
}
__global__ void fill_f32_k(float* p, float v, int n) {
  int t = blockIdx.x * blockDim.x + threadIdx.x;
  if (t < n) p[t] = v;
}

// --------------------------------------------- stage 1: LN1 + q/k/v/skip proj
__global__ __launch_bounds__(128) void proj_k(
    const float* __restrict__ x, const float* __restrict__ g, const float* __restrict__ bln,
    const float* __restrict__ wq, const float* __restrict__ bq,
    const float* __restrict__ wk, const float* __restrict__ bk,
    const float* __restrict__ wv, const float* __restrict__ bv,
    const float* __restrict__ ws, const float* __restrict__ bs,
    _Float16* __restrict__ qo, _Float16* __restrict__ ko, _Float16* __restrict__ vo,
    float* __restrict__ skipo)
{
  int r = blockIdx.x;          // (b*T + t)*N + n
  int j = threadIdx.x;         // output column 0..127
  const float* xr = x + (size_t)r * FIN_;
  float xs[FIN_], ln[FIN_];
  float mu = 0.f;
#pragma unroll
  for (int c = 0; c < FIN_; ++c) { xs[c] = xr[c]; mu += xs[c]; }
  mu *= (1.f / FIN_);
  float var = 0.f;
#pragma unroll
  for (int c = 0; c < FIN_; ++c) { float d = xs[c] - mu; var += d * d; }
  var *= (1.f / FIN_);
  float inv = rsqrtf(var + 1e-5f);
#pragma unroll
  for (int c = 0; c < FIN_; ++c) ln[c] = (xs[c] - mu) * inv * g[c] + bln[c];

  float aq = bq[j], ak = bk[j], av = bv[j], as = bs[j];
#pragma unroll
  for (int c = 0; c < FIN_; ++c) {
    float l = ln[c], xv = xs[c];
    aq += l * wq[c * HID_ + j];
    ak += l * wk[c * HID_ + j];
    av += l * wv[c * HID_ + j];
    as += xv * ws[c * HID_ + j];
  }
  const float scale = 0.08838834764831845f;   // 1/sqrt(128), folded into q
  size_t qi = (size_t)r * HID_ + j;
  qo[qi] = (_Float16)(aq * scale);
  ko[qi] = (_Float16)ak;
  vo[qi] = (_Float16)av;
  int bt = r / N_, n = r % N_, b = bt / T_, t = bt % T_;
  skipo[(((size_t)b * N_ + n) * T_ + t) * HID_ + j] = as;
}

// --------------------------------------------- stage 2: causal flash attention
// grid: (N_/16, B_*T_), 1 wave per block. q/k/v: [bt][n][128] f16, q pre-scaled.
// K/V tiles double-buffered in LDS via async global->LDS copies.
__global__ __launch_bounds__(32) void flash_k(
    const _Float16* __restrict__ q, const _Float16* __restrict__ k,
    const _Float16* __restrict__ v, _Float16* __restrict__ ctx)
{
  __shared__ __align__(16) _Float16 sQ[16 * HID_];
  __shared__ __align__(16) _Float16 sK[2][32 * HID_];
  __shared__ __align__(16) _Float16 sV[2][32 * HID_];
  __shared__ float    sS[16 * 32];
  __shared__ __align__(16) _Float16 sP[16 * 32];
  __shared__ float rowm[16], rowl[16], rowa[16];

  int lane = threadIdx.x;
  int qt = blockIdx.x, bt = blockIdx.y;
  const _Float16* qb = q + ((size_t)bt * N_ + qt * 16) * HID_;
#pragma unroll 1
  for (int idx = lane; idx < 16 * HID_ / 8; idx += 32)
    ((uint4*)sQ)[idx] = ((const uint4*)qb)[idx];
  if (lane < 16) { rowm[lane] = -INFINITY; rowl[lane] = 0.f; }

  const _Float16* kbase = k + (size_t)bt * N_ * HID_;
  const _Float16* vbase = v + (size_t)bt * N_ * HID_;
  unsigned ldsK0 = (unsigned)(uintptr_t)&sK[0][0];
  unsigned ldsK1 = (unsigned)(uintptr_t)&sK[1][0];
  unsigned ldsV0 = (unsigned)(uintptr_t)&sV[0][0];
  unsigned ldsV1 = (unsigned)(uintptr_t)&sV[1][0];

  auto stage = [&](int j, int bb) {
    const uint4* kb = (const uint4*)(kbase + (size_t)j * 32 * HID_);
    const uint4* vb = (const uint4*)(vbase + (size_t)j * 32 * HID_);
    unsigned lk = bb ? ldsK1 : ldsK0;
    unsigned lv = bb ? ldsV1 : ldsV0;
#pragma unroll
    for (int it = 0; it < 16; ++it) {
      unsigned idx = (unsigned)(lane + it * 32);
      async_cp16(lk + idx * 16u, (unsigned long long)(uintptr_t)(kb + idx));
      async_cp16(lv + idx * 16u, (unsigned long long)(uintptr_t)(vb + idx));
    }
  };

  int jmax = (qt * 16 + 15) / 32;
  stage(0, 0);                 // prefetch first K/V tile (32 async ops in flight)
  __syncthreads();

  v16h qa[4];
#pragma unroll
  for (int kc = 0; kc < 4; ++kc) qa[kc] = load_a(sQ + kc * 32, HID_);

  v8f o[8];
#pragma unroll
  for (int d = 0; d < 8; ++d) o[d] = zero8();

  for (int j = 0; j <= jmax; ++j) {
    int cur = j & 1;
    if (j < jmax) {
      stage(j + 1, cur ^ 1);   // prefetch next tile into the other buffer
      WAIT_ASYNC("32");        // current tile done; only next tile outstanding
    } else {
      WAIT_ASYNC("0");
    }
    __syncthreads();
    const _Float16* cK = &sK[cur][0];
    const _Float16* cV = &sV[cur][0];

    // scores S = Q * K^T  -> two 16x16 tiles
#pragma unroll
    for (int nt = 0; nt < 2; ++nt) {
      v8f sc = zero8();
#pragma unroll
      for (int kc = 0; kc < 4; ++kc) {
        v16h bf = load_bt(cK + (nt * 16) * HID_ + kc * 32, HID_);
        sc = wmma16(qa[kc], bf, sc);
      }
      int n = lane & 15, rhi = (lane >> 4) << 3;
#pragma unroll
      for (int r = 0; r < 8; ++r) sS[(rhi + r) * 32 + nt * 16 + n] = sc[r];
    }
    __syncthreads();

    // online softmax: 2 lanes per row (16 cols each), shfl combine
    {
      int m = lane & 15, hp = lane >> 4;
      int grow = qt * 16 + m;
      float mx = rowm[m];
      float vals[16];
      float tm = -INFINITY;
#pragma unroll
      for (int p = 0; p < 16; ++p) {
        int col = hp * 16 + p;
        float sv = (j * 32 + col <= grow) ? sS[m * 32 + col] : -INFINITY;
        vals[p] = sv;
        tm = fmaxf(tm, sv);
      }
      tm = fmaxf(tm, __shfl_xor(tm, 16));
      float nm = fmaxf(mx, tm);
      float sum = 0.f;
#pragma unroll
      for (int p = 0; p < 16; ++p) {
        float e = __expf(vals[p] - nm);
        sP[m * 32 + hp * 16 + p] = (_Float16)e;
        sum += e;
      }
      sum += __shfl_xor(sum, 16);
      if (lane < 16) {
        float al = __expf(mx - nm);   // mx==-inf -> 0
        rowl[m] = rowl[m] * al + sum;
        rowm[m] = nm;
        rowa[m] = al;
      }
    }
    __syncthreads();

    // rescale accumulators, then O += P * V
    {
      int rhi = (lane >> 4) << 3;
#pragma unroll
      for (int r = 0; r < 8; ++r) {
        float al = rowa[rhi + r];
#pragma unroll
        for (int d = 0; d < 8; ++d) o[d][r] *= al;
      }
    }
    v16h pa = load_a(sP, 32);
#pragma unroll
    for (int d = 0; d < 8; ++d) {
      v16h bf = load_b(cV + d * 16, HID_);
      o[d] = wmma16(pa, bf, o[d]);
    }
    __syncthreads();
  }

  // epilogue: normalize and scatter to [b][n][t][128]
  int b = bt >> 3, t = bt & 7;
  int col = lane & 15, rhi = (lane >> 4) << 3;
#pragma unroll
  for (int r = 0; r < 8; ++r) {
    int n = qt * 16 + rhi + r;
    float invl = 1.f / rowl[rhi + r];
    size_t base = (((size_t)b * N_ + n) * T_ + t) * HID_;
#pragma unroll
    for (int d = 0; d < 8; ++d)
      ctx[base + d * 16 + col] = (_Float16)(o[d][r] * invl);
  }
}

// --------------------------------------------- generic WMMA row GEMM
// out[row][n] = act( A[row][:] @ W + bias ) (+ res); block = 4 waves * 16 rows
template <int K, int N, int ACT, bool RES, bool OUTF16>
__global__ __launch_bounds__(128) void gemm_k(
    const _Float16* __restrict__ A, const _Float16* __restrict__ W,
    const float* __restrict__ bias, const float* __restrict__ res,
    void* __restrict__ outp)
{
  __shared__ __align__(16) _Float16 sA[4][16 * K];
  __shared__ __align__(16) _Float16 sW[32 * N];
  int wave = threadIdx.x >> 5, lane = threadIdx.x & 31;
  int row0 = blockIdx.x * 64 + wave * 16;
  const _Float16* Arow = A + (size_t)row0 * K;
#pragma unroll 1
  for (int idx = lane; idx < 16 * K / 8; idx += 32)
    ((uint4*)sA[wave])[idx] = ((const uint4*)Arow)[idx];

  v8f acc[N / 16];
#pragma unroll
  for (int i = 0; i < N / 16; ++i) acc[i] = zero8();

  for (int kc = 0; kc < K; kc += 32) {
    __syncthreads();
#pragma unroll 1
    for (int idx = threadIdx.x; idx < 32 * N / 8; idx += 128)
      ((uint4*)sW)[idx] = ((const uint4*)(W + (size_t)kc * N))[idx];
    if (kc + 32 < K) __builtin_prefetch(W + (size_t)(kc + 32) * N, 0, 1);
    __syncthreads();
    v16h a = load_a(&sA[wave][kc], K);
#pragma unroll
    for (int nt = 0; nt < N / 16; ++nt) {
      v16h b = load_b(sW + nt * 16, N);
      acc[nt] = wmma16(a, b, acc[nt]);
    }
  }

  int col = lane & 15, rhi = (lane >> 4) << 3;
#pragma unroll
  for (int nt = 0; nt < N / 16; ++nt) {
    int n = nt * 16 + col;
    float bn = bias ? bias[n] : 0.f;
#pragma unroll
    for (int r = 0; r < 8; ++r) {
      int row = row0 + rhi + r;
      float v = acc[nt][r] + bn;
      if (RES) v += res[(size_t)row * N + n];
      if (ACT == 1) v = v > 0.f ? v : 0.f;
      if (OUTF16) ((_Float16*)outp)[(size_t)row * N + n] = (_Float16)v;
      else        ((float*)outp)[(size_t)row * N + n] = v;
    }
  }
}

// --------------------------------------------- LayerNorm over 128 (f32 -> f16)
__global__ __launch_bounds__(128) void ln128_k(
    const float* __restrict__ in, const float* __restrict__ g,
    const float* __restrict__ b, _Float16* __restrict__ out)
{
  __shared__ float red[128];
  int r = blockIdx.x, j = threadIdx.x;
  float v = in[(size_t)r * HID_ + j];
  red[j] = v; __syncthreads();
  for (int s = 64; s > 0; s >>= 1) { if (j < s) red[j] += red[j + s]; __syncthreads(); }
  float mu = red[0] * (1.f / HID_);
  __syncthreads();
  float d = v - mu;
  red[j] = d * d; __syncthreads();
  for (int s = 64; s > 0; s >>= 1) { if (j < s) red[j] += red[j + s]; __syncthreads(); }
  float var = red[0] * (1.f / HID_);
  out[(size_t)r * HID_ + j] = (_Float16)(d * rsqrtf(var + 1e-5f) * g[j] + b[j]);
}

// --------------------------------------------- node attention (8x8, no mask)
__global__ __launch_bounds__(32) void nodeattn_k(
    const _Float16* __restrict__ q, const _Float16* __restrict__ k,
    const _Float16* __restrict__ v, _Float16* __restrict__ ctx)
{
  __shared__ __align__(16) _Float16 sq[8 * HID_], sk[8 * HID_], sv[8 * HID_];
  __shared__ float sa[8][8];
  int lane = threadIdx.x;
  size_t base = (size_t)blockIdx.x * 8 * HID_;
#pragma unroll 1
  for (int idx = lane; idx < 8 * HID_ / 8; idx += 32) {
    ((uint4*)sq)[idx] = ((const uint4*)(q + base))[idx];
    ((uint4*)sk)[idx] = ((const uint4*)(k + base))[idx];
    ((uint4*)sv)[idx] = ((const uint4*)(v + base))[idx];
  }
  __syncthreads();
  const float scale = 0.08838834764831845f;
#pragma unroll 1
  for (int p = lane; p < 64; p += 32) {
    int i = p >> 3, jj = p & 7;
    float s = 0.f;
    for (int d = 0; d < HID_; ++d) s += (float)sq[i * HID_ + d] * (float)sk[jj * HID_ + d];
    sa[i][jj] = s * scale;
  }
  __syncthreads();
  if (lane < 8) {
    int i = lane;
    float mx = -INFINITY;
#pragma unroll
    for (int jj = 0; jj < 8; ++jj) mx = fmaxf(mx, sa[i][jj]);
    float sum = 0.f;
#pragma unroll
    for (int jj = 0; jj < 8; ++jj) { float e = __expf(sa[i][jj] - mx); sa[i][jj] = e; sum += e; }
    float inv = 1.f / sum;
#pragma unroll
    for (int jj = 0; jj < 8; ++jj) sa[i][jj] *= inv;
  }
  __syncthreads();
#pragma unroll
  for (int i = 0; i < 8; ++i)
    for (int d = lane; d < HID_; d += 32) {
      float s = 0.f;
#pragma unroll
      for (int jj = 0; jj < 8; ++jj) s += sa[i][jj] * (float)sv[jj * HID_ + d];
      ctx[base + i * HID_ + d] = (_Float16)s;
    }
}

// --------------------------------------------- GAT pieces
__global__ void gather_last_k(const float* __restrict__ h3, _Float16* __restrict__ out) {
  int t = blockIdx.x * blockDim.x + threadIdx.x;
  if (t >= NODES_ * HID_) return;
  int i = t / HID_, j = t % HID_;
  out[t] = (_Float16)h3[(((size_t)i * T_) + (T_ - 1)) * HID_ + j];
}
__global__ void al_k(const float* __restrict__ hw, const float* __restrict__ asrc,
                     const float* __restrict__ adst, int H, int ch,
                     float* __restrict__ als, float* __restrict__ ald) {
  int t = blockIdx.x * blockDim.x + threadIdx.x;
  if (t >= NODES_ * H) return;
  int i = t / H, h = t % H;
  float s = 0.f, d = 0.f;
  for (int c = 0; c < ch; ++c) {
    float v = hw[(size_t)i * H * ch + h * ch + c];
    s += v * asrc[h * ch + c];
    d += v * adst[h * ch + c];
  }
  als[t] = s; ald[t] = d;
}
__device__ __forceinline__ void edge_decode(const int* ei, int e, int& src, int& dst, bool& drop) {
  if (e < B_ * E_) {
    int b = e / E_, le = e % E_;
    int s0 = ei[le], d0 = ei[E_ + le];
    drop = (s0 == d0);
    src = s0 + b * N_; dst = d0 + b * N_;
  } else {
    int i = e - B_ * E_;
    src = dst = i; drop = false;
  }
}
__global__ void edge_max_k(const int* __restrict__ ei, const float* __restrict__ als,
                           const float* __restrict__ ald, int H, unsigned* __restrict__ amax) {
  int t = blockIdx.x * blockDim.x + threadIdx.x;
  int total = (B_ * E_ + NODES_) * H;
  if (t >= total) return;
  int e = t / H, h = t % H, src, dst; bool drop;
  edge_decode(ei, e, src, dst, drop);
  if (drop) return;
  float a = als[src * H + h] + ald[dst * H + h];
  a = a > 0.f ? a : 0.2f * a;
  atomicMax(&amax[dst * H + h], f2ord(a));
}
__global__ void edge_agg_k(const int* __restrict__ ei, const float* __restrict__ als,
                           const float* __restrict__ ald, int H, int ch,
                           const unsigned* __restrict__ amax, const float* __restrict__ hw,
                           float* __restrict__ denom, float* __restrict__ agg) {
  int CHT = H * ch;
  int t = blockIdx.x * blockDim.x + threadIdx.x;
  int total = (B_ * E_ + NODES_) * CHT;
  if (t >= total) return;
  int e = t / CHT, c = t % CHT, h = c / ch, src, dst; bool drop;
  edge_decode(ei, e, src, dst, drop);
  if (drop) return;
  float a = als[src * H + h] + ald[dst * H + h];
  a = a > 0.f ? a : 0.2f * a;
  float ex = __expf(a - ord2f(amax[dst * H + h]));
  if (c == h * ch) atomicAdd(&denom[dst * H + h], ex);
  atomicAdd(&agg[(size_t)dst * CHT + c], ex * hw[(size_t)src * CHT + c]);
}
__global__ void fin1_k(const float* __restrict__ agg, const float* __restrict__ denom,
                       const float* __restrict__ bias, _Float16* __restrict__ out) {
  int t = blockIdx.x * blockDim.x + threadIdx.x;
  if (t >= NODES_ * HID_) return;
  int i = t / HID_, j = t % HID_, h = j / 32;
  float v = agg[t] / denom[i * 4 + h] + bias[j];
  v = v > 0.f ? v : __expf(v) - 1.f;  // ELU
  out[t] = (_Float16)v;
}
__global__ void fin2_k(const float* __restrict__ agg, const float* __restrict__ denom,
                       const float* __restrict__ bias, float* __restrict__ out) {
  int t = blockIdx.x * blockDim.x + threadIdx.x;
  if (t >= NODES_ * 64) return;
  int i = t / 64, j = t % 64;
  out[t] = agg[t] / denom[i] + bias[j];
}
__global__ void ffn_k(const float* __restrict__ g2, const float* __restrict__ w,
                      const float* __restrict__ b, float* __restrict__ out) {
  int t = blockIdx.x * blockDim.x + threadIdx.x;
  if (t >= NODES_ * 14) return;
  int f = t % 14, bn = t / 14, bb = bn / N_, n = bn % N_;
  float s = b[f];
  for (int c = 0; c < 64; ++c) s += g2[(size_t)bn * 64 + c] * w[c * 14 + f];
  out[((size_t)bb * 14 + f) * N_ + n] = s;
}

// ================================================================= launcher
extern "C" void kernel_launch(void* const* d_in, const int* in_sizes, int n_in,
                              void* d_out, int out_size, void* d_ws, size_t ws_size,
                              hipStream_t stream) {
  (void)in_sizes; (void)n_in; (void)out_size; (void)ws_size;
  const float* x      = (const float*)d_in[0];
  const int*   ei     = (const int*)d_in[1];
  const float* ln1_g  = (const float*)d_in[2];
  const float* ln1_b  = (const float*)d_in[3];
  const float* t_wq   = (const float*)d_in[4];
  const float* t_bq   = (const float*)d_in[5];
  const float* t_wk   = (const float*)d_in[6];
  const float* t_bk   = (const float*)d_in[7];
  const float* t_wv   = (const float*)d_in[8];
  const float* t_bv   = (const float*)d_in[9];
  const float* t_wo   = (const float*)d_in[10];
  const float* t_bo   = (const float*)d_in[11];
  const float* skip_w = (const float*)d_in[12];
  const float* skip_b = (const float*)d_in[13];
  const float* ln2_g  = (const float*)d_in[14];
  const float* ln2_b  = (const float*)d_in[15];
  const float* s_wq   = (const float*)d_in[16];
  const float* s_bq   = (const float*)d_in[17];
  const float* s_wk   = (const float*)d_in[18];
  const float* s_bk   = (const float*)d_in[19];
  const float* s_wv   = (const float*)d_in[20];
  const float* s_bv   = (const float*)d_in[21];
  const float* s_wo   = (const float*)d_in[22];
  const float* s_bo   = (const float*)d_in[23];
  const float* ln3_g  = (const float*)d_in[24];
  const float* ln3_b  = (const float*)d_in[25];
  const float* mlp_w1 = (const float*)d_in[26];
  const float* mlp_b1 = (const float*)d_in[27];
  const float* mlp_w2 = (const float*)d_in[28];
  const float* mlp_b2 = (const float*)d_in[29];
  const float* gat1_w = (const float*)d_in[30];
  const float* gat1_as= (const float*)d_in[31];
  const float* gat1_ad= (const float*)d_in[32];
  const float* gat1_b = (const float*)d_in[33];
  const float* gat2_w = (const float*)d_in[34];
  const float* gat2_as= (const float*)d_in[35];
  const float* gat2_ad= (const float*)d_in[36];
  const float* gat2_b = (const float*)d_in[37];
  const float* ffn_w  = (const float*)d_in[38];
  const float* ffn_b  = (const float*)d_in[39];
  float* outp = (float*)d_out;

  // workspace carve-up
  char* base = (char*)d_ws;
  size_t off = 0;
  auto alloc = [&](size_t bytes) { size_t o = off; off = (off + bytes + 255) & ~(size_t)255; return (void*)(base + o); };
  _Float16* q16   = (_Float16*)alloc((size_t)ROWS_ * HID_ * 2);
  _Float16* k16   = (_Float16*)alloc((size_t)ROWS_ * HID_ * 2);
  _Float16* v16   = (_Float16*)alloc((size_t)ROWS_ * HID_ * 2);
  _Float16* ctx16 = (_Float16*)alloc((size_t)ROWS_ * HID_ * 2);
  _Float16* z16   = (_Float16*)alloc((size_t)ROWS_ * HID_ * 2);
  _Float16* m116  = (_Float16*)alloc((size_t)ROWS_ * 256 * 2);
  float* skip32   = (float*)alloc((size_t)ROWS_ * HID_ * 4);
  float* h1       = (float*)alloc((size_t)ROWS_ * HID_ * 4);
  float* h2       = (float*)alloc((size_t)ROWS_ * HID_ * 4);
  float* h3       = (float*)alloc((size_t)ROWS_ * HID_ * 4);
  _Float16* w16   = (_Float16*)alloc(172032 * 2);      // converted weights arena
  _Float16* hsel16= (_Float16*)alloc((size_t)NODES_ * HID_ * 2);
  float* hw1      = (float*)alloc((size_t)NODES_ * HID_ * 4);
  float* als1     = (float*)alloc((size_t)NODES_ * 4 * 4);
  float* ald1     = (float*)alloc((size_t)NODES_ * 4 * 4);
  unsigned* amax1 = (unsigned*)alloc((size_t)NODES_ * 4 * 4);
  float* den1     = (float*)alloc((size_t)NODES_ * 4 * 4);
  float* agg1     = (float*)alloc((size_t)NODES_ * HID_ * 4);
  _Float16* g116  = (_Float16*)alloc((size_t)NODES_ * HID_ * 2);
  float* hw2      = (float*)alloc((size_t)NODES_ * 64 * 4);
  float* als2     = (float*)alloc((size_t)NODES_ * 4);
  float* ald2     = (float*)alloc((size_t)NODES_ * 4);
  unsigned* amax2 = (unsigned*)alloc((size_t)NODES_ * 4);
  float* den2     = (float*)alloc((size_t)NODES_ * 4);
  float* agg2     = (float*)alloc((size_t)NODES_ * 64 * 4);
  float* g2out    = (float*)alloc((size_t)NODES_ * 64 * 4);

  _Float16* wo16  = w16 + 0;        // 128x128
  _Float16* swq16 = w16 + 16384;    // 128x128
  _Float16* swk16 = w16 + 32768;
  _Float16* swv16 = w16 + 49152;
  _Float16* swo16 = w16 + 65536;
  _Float16* mw116 = w16 + 81920;    // 128x256
  _Float16* mw216 = w16 + 114688;   // 256x128
  _Float16* g1w16 = w16 + 147456;   // 128x128
  _Float16* g2w16 = w16 + 163840;   // 128x64

  auto cvt = [&](const float* s, _Float16* d, int n) {
    cvt_f32_f16_k<<<(n + 255) / 256, 256, 0, stream>>>(s, d, n);
  };
  cvt(t_wo, wo16, 16384);   cvt(s_wq, swq16, 16384);  cvt(s_wk, swk16, 16384);
  cvt(s_wv, swv16, 16384);  cvt(s_wo, swo16, 16384);  cvt(mlp_w1, mw116, 32768);
  cvt(mlp_w2, mw216, 32768);cvt(gat1_w, g1w16, 16384);cvt(gat2_w, g2w16, 8192);

  // stage 1: LN1 + q/k/v/skip projection
  proj_k<<<ROWS_, 128, 0, stream>>>(x, ln1_g, ln1_b, t_wq, t_bq, t_wk, t_bk,
                                    t_wv, t_bv, skip_w, skip_b, q16, k16, v16, skip32);
  // stage 2: flash attention over the 2048 axis, per (b,t)
  flash_k<<<dim3(N_ / 16, B_ * T_), 32, 0, stream>>>(q16, k16, v16, ctx16);
  // h1 = skip + ctx @ t_wo + t_bo
  gemm_k<128, 128, 0, true, false><<<ROWS_ / 64, 128, 0, stream>>>(ctx16, wo16, t_bo, skip32, h1);
  // z = LN2(h1); q2/k2/v2
  ln128_k<<<ROWS_, 128, 0, stream>>>(h1, ln2_g, ln2_b, z16);
  gemm_k<128, 128, 0, false, true><<<ROWS_ / 64, 128, 0, stream>>>(z16, swq16, s_bq, nullptr, q16);
  gemm_k<128, 128, 0, false, true><<<ROWS_ / 64, 128, 0, stream>>>(z16, swk16, s_bk, nullptr, k16);
  gemm_k<128, 128, 0, false, true><<<ROWS_ / 64, 128, 0, stream>>>(z16, swv16, s_bv, nullptr, v16);
  // 8x8 node attention per (b,n)
  nodeattn_k<<<NODES_, 32, 0, stream>>>(q16, k16, v16, ctx16);
  // h2 = h1 + ctx2 @ s_wo + s_bo
  gemm_k<128, 128, 0, true, false><<<ROWS_ / 64, 128, 0, stream>>>(ctx16, swo16, s_bo, h1, h2);
  // MLP: h3 = h2 + relu(LN3(h2) @ w1 + b1) @ w2 + b2
  ln128_k<<<ROWS_, 128, 0, stream>>>(h2, ln3_g, ln3_b, z16);
  gemm_k<128, 256, 1, false, true><<<ROWS_ / 64, 128, 0, stream>>>(z16, mw116, mlp_b1, nullptr, m116);
  gemm_k<256, 128, 0, true, false><<<ROWS_ / 64, 128, 0, stream>>>(m116, mw216, mlp_b2, h2, h3);

  // GAT1
  gather_last_k<<<(NODES_ * HID_ + 255) / 256, 256, 0, stream>>>(h3, hsel16);
  gemm_k<128, 128, 0, false, false><<<NODES_ / 64, 128, 0, stream>>>(hsel16, g1w16, nullptr, nullptr, hw1);
  al_k<<<(NODES_ * 4 + 255) / 256, 256, 0, stream>>>(hw1, gat1_as, gat1_ad, 4, 32, als1, ald1);
  {
    unsigned ninf = ~0xFF800000u;  // f2ord(-inf)
    fill_u32_k<<<(NODES_ * 4 + 255) / 256, 256, 0, stream>>>(amax1, ninf, NODES_ * 4);
    fill_f32_k<<<(NODES_ * 4 + 255) / 256, 256, 0, stream>>>(den1, 0.f, NODES_ * 4);
    fill_f32_k<<<(NODES_ * HID_ + 255) / 256, 256, 0, stream>>>(agg1, 0.f, NODES_ * HID_);
  }
  {
    int tot = (B_ * E_ + NODES_) * 4;
    edge_max_k<<<(tot + 255) / 256, 256, 0, stream>>>(ei, als1, ald1, 4, amax1);
    int tot2 = (B_ * E_ + NODES_) * 128;
    edge_agg_k<<<(tot2 + 255) / 256, 256, 0, stream>>>(ei, als1, ald1, 4, 32, amax1, hw1, den1, agg1);
  }
  fin1_k<<<(NODES_ * HID_ + 255) / 256, 256, 0, stream>>>(agg1, den1, gat1_b, g116);

  // GAT2
  gemm_k<128, 64, 0, false, false><<<NODES_ / 64, 128, 0, stream>>>(g116, g2w16, nullptr, nullptr, hw2);
  al_k<<<(NODES_ + 255) / 256, 256, 0, stream>>>(hw2, gat2_as, gat2_ad, 1, 64, als2, ald2);
  {
    unsigned ninf = ~0xFF800000u;
    fill_u32_k<<<(NODES_ + 255) / 256, 256, 0, stream>>>(amax2, ninf, NODES_);
    fill_f32_k<<<(NODES_ + 255) / 256, 256, 0, stream>>>(den2, 0.f, NODES_);
    fill_f32_k<<<(NODES_ * 64 + 255) / 256, 256, 0, stream>>>(agg2, 0.f, NODES_ * 64);
  }
  {
    int tot = (B_ * E_ + NODES_);
    edge_max_k<<<(tot + 255) / 256, 256, 0, stream>>>(ei, als2, ald2, 1, amax2);
    int tot2 = (B_ * E_ + NODES_) * 64;
    edge_agg_k<<<(tot2 + 255) / 256, 256, 0, stream>>>(ei, als2, ald2, 1, 64, amax2, hw2, den2, agg2);
  }
  fin2_k<<<(NODES_ * 64 + 255) / 256, 256, 0, stream>>>(agg2, den2, gat2_b, g2out);

  // final FFN + transpose to [B, 14, N]
  ffn_k<<<(NODES_ * 14 + 255) / 256, 256, 0, stream>>>(g2out, ffn_w, ffn_b, outp);
}
